// EdgeModel_86535001080076
// MI455X (gfx1250) — matrix operand order repaired and checked by reading, not compiled, not run on hardware
//
#include <hip/hip_runtime.h>
#include <hip/hip_bf16.h>

// EdgeModel MLP: out = ELU(concat(src,dest,edge_attr) @ W1 + b1) @ W2 + b2
// E=200000, FN=512, FE=256, IN=1280, HIDDEN=32, OUT=128. All fp32.
// HBM-bound (~1.13 GB @ 23.3 TB/s ~ 49us floor); compute via f16 WMMA.

typedef __attribute__((ext_vector_type(16))) _Float16 v16h;
typedef __attribute__((ext_vector_type(8)))  float    v8f;
typedef __attribute__((ext_vector_type(4)))  float    f4;

#define WAVES_PER_BLOCK 8
#define BLOCK_THREADS   (WAVES_PER_BLOCK * 32)

// ---- shared-memory fragment blobs -------------------------------------------
// sW1: 40 k-steps x 2 n-tiles x 32 lanes x 16 halfs  = 40960 f16 = 80 KB
// sW2:              8 n-tiles x 32 lanes x 16 halfs  =  4096 f16 =  8 KB
// sH : per-wave 16x32 f32 scratch for D->A relayout  = 16 KB
//
// Assumed WMMA f16 fragment layouts (wave32):
//  A 16x32 (doc 7.12.2): lane<16: M=lane,    K = {0..7} in V0-3, {16..23} in V4-7
//                        lane>=16: M=lane-16, K = {8..15},        {24..31}
//  B 32x16 (from sparse-B 64x16 pattern):  N = lane&15,
//                        lane<16: K = {0..7,16..23}, lane>=16: K = {8..15,24..31} (same order as A)
//  C/D 16x16 f32 (doc): VGPR r: lanes 0-15 -> (M=r, N=lane); lanes 16-31 -> (M=8+r, N=lane-16)

// One K=32 step of GEMM1: stream 16 fp32 from the edge row, cvt to f16 A-frag,
// multiply against two pre-swizzled W1 B-frags (hidden 0-15 and 16-31).
__device__ __forceinline__ void kstep(const float* __restrict__ p,
                                      const _Float16* __restrict__ w,
                                      v8f& acc0, v8f& acc1) {
  const f4 x0 = __builtin_nontemporal_load((const f4*)(p));
  const f4 x1 = __builtin_nontemporal_load((const f4*)(p + 4));
  const f4 x2 = __builtin_nontemporal_load((const f4*)(p + 16));
  const f4 x3 = __builtin_nontemporal_load((const f4*)(p + 20));
  v16h a;
#pragma unroll
  for (int j = 0; j < 4; ++j) {
    a[j]      = (_Float16)x0[j];
    a[4 + j]  = (_Float16)x1[j];
    a[8 + j]  = (_Float16)x2[j];
    a[12 + j] = (_Float16)x3[j];
  }
  const v16h bf0 = *(const v16h*)(w);            // n-tile 0 (hidden 0..15)
  const v16h bf1 = *(const v16h*)(w + 512);      // n-tile 1 (hidden 16..31)
  acc0 = __builtin_amdgcn_wmma_f32_16x16x32_f16(false, a, false, bf0,
                                                (short)0, acc0, false, false);
  acc1 = __builtin_amdgcn_wmma_f32_16x16x32_f16(false, a, false, bf1,
                                                (short)0, acc1, false, false);
}

__global__ __launch_bounds__(BLOCK_THREADS) void edge_mlp_kernel(
    const float* __restrict__ src, const float* __restrict__ dst,
    const float* __restrict__ eattr,
    const float* __restrict__ W1, const float* __restrict__ b1,
    const float* __restrict__ W2, const float* __restrict__ b2,
    float* __restrict__ out, int E) {

  __shared__ __align__(32) _Float16 sW1[40 * 2 * 32 * 16];
  __shared__ __align__(32) _Float16 sW2[8 * 32 * 16];
  __shared__ __align__(16) float    sH[WAVES_PER_BLOCK][16][32];

  const int tid = threadIdx.x;

  // ---- cooperative pre-swizzle of W1/W2 into B-fragment order (f32 -> f16) ---
  for (int idx = tid; idx < 40 * 2 * 32 * 16; idx += BLOCK_THREADS) {
    const int j    = idx & 15;
    const int lane = (idx >> 4) & 31;
    const int t    = (idx >> 9) & 1;
    const int s    = idx >> 10;
    const int k    = s * 32 + ((lane < 16) ? 0 : 8) + j + (j & 8);   // 0..1279
    const int n    = t * 16 + (lane & 15);                  // 0..31
    sW1[idx] = (_Float16)W1[k * 32 + n];
  }
  for (int idx = tid; idx < 8 * 32 * 16; idx += BLOCK_THREADS) {
    const int j    = idx & 15;
    const int lane = (idx >> 4) & 31;
    const int t    = idx >> 9;
    const int k    = ((lane < 16) ? 0 : 8) + j + (j & 8);            // 0..31
    const int n    = t * 16 + (lane & 15);                  // 0..127
    sW2[idx] = (_Float16)W2[k * 128 + n];
  }
  __syncthreads();

  const int lane  = tid & 31;
  const int wave  = tid >> 5;
  const int gwave = blockIdx.x * WAVES_PER_BLOCK + wave;
  const int nwav  = gridDim.x * WAVES_PER_BLOCK;
  const int n0    = lane & 15;            // column id within fragment
  const int mb    = (lane < 16) ? 0 : 8;  // D-layout row base
  const int koff  = (lane < 16) ? 0 : 8;  // A-layout K base

  // hoisted per-lane biases
  const float b1lo = b1[n0];
  const float b1hi = b1[n0 + 16];
  float b2v[8];
#pragma unroll
  for (int t = 0; t < 8; ++t) b2v[t] = b2[t * 16 + n0];  // feature = t*16 + n0

  const _Float16* __restrict__ wlane = &sW1[lane * 16];
  const int ntiles = (E + 15) >> 4;

  for (int tile = gwave; tile < ntiles; tile += nwav) {
    int e = tile * 16 + n0;                 // this lane's A-matrix row (edge)
    if (e >= E) e = E - 1;                  // clamp for ragged tail (loads only)
    const float* __restrict__ pS = src   + (size_t)e * 512 + koff;
    const float* __restrict__ pD = dst   + (size_t)e * 512 + koff;
    const float* __restrict__ pA = eattr + (size_t)e * 256 + koff;

    v8f acc0 = {};
    v8f acc1 = {};

    // -------- GEMM1: [16 x 1280] x [1280 x 32]; branch-free segmented K loop --
#pragma unroll
    for (int s = 0; s < 16; ++s)            // k = 0..511  (src)
      kstep(pS + s * 32, wlane + s * 1024, acc0, acc1);
#pragma unroll
    for (int s = 0; s < 16; ++s)            // k = 512..1023 (dest)
      kstep(pD + s * 32, wlane + (16 + s) * 1024, acc0, acc1);
#pragma unroll
    for (int s = 0; s < 8; ++s)             // k = 1024..1279 (edge_attr)
      kstep(pA + s * 32, wlane + (32 + s) * 1024, acc0, acc1);

    // ---------------- bias + ELU, stash h into per-wave LDS ------------------
#pragma unroll
    for (int r = 0; r < 8; ++r) {
      float v0 = acc0[r] + b1lo;            // hidden n0
      float v1 = acc1[r] + b1hi;            // hidden n0+16
      v0 = (v0 > 0.0f) ? v0 : (__expf(v0) - 1.0f);
      v1 = (v1 > 0.0f) ? v1 : (__expf(v1) - 1.0f);
      sH[wave][mb + r][n0]      = v0;       // row = edge-in-tile, col = hidden
      sH[wave][mb + r][n0 + 16] = v1;
    }
    __asm__ volatile("" ::: "memory");      // keep same-wave LDS store->load order

    // ---------------- relayout D -> A for GEMM2 (same-wave LDS roundtrip) ----
    const float* hrow = &sH[wave][n0][0];   // A row M = n0
    const f4 h0 = *(const f4*)(hrow + koff);
    const f4 h1 = *(const f4*)(hrow + koff + 4);
    const f4 h2 = *(const f4*)(hrow + koff + 16);
    const f4 h3 = *(const f4*)(hrow + koff + 20);
    v16h a2;
#pragma unroll
    for (int j = 0; j < 4; ++j) {
      a2[j]      = (_Float16)h0[j];
      a2[4 + j]  = (_Float16)h1[j];
      a2[8 + j]  = (_Float16)h2[j];
      a2[12 + j] = (_Float16)h3[j];
    }

    // ---------------- GEMM2: [16 x 32] x [32 x 128] = 8 WMMA tiles -----------
    const int ebase = tile * 16 + mb;
    const bool full_tile = (tile * 16 + 16 <= E);   // branch hoisted out of stores
    float* __restrict__ outbase = out + (size_t)ebase * 128 + n0;
#pragma unroll
    for (int t = 0; t < 8; ++t) {
      const v16h bw = *(const v16h*)&sW2[(t * 32 + lane) * 16];
      v8f c = {};
      const v8f d = __builtin_amdgcn_wmma_f32_16x16x32_f16(false, a2, false, bw,
                                                           (short)0, c, false, false);
      if (full_tile) {
#pragma unroll
        for (int r = 0; r < 8; ++r)
          // edge = ebase+r, feature = t*16+n0 -> coalesced across lanes
          __builtin_nontemporal_store(d[r] + b2v[t],
                                      outbase + (size_t)r * 128 + t * 16);
      } else {
#pragma unroll
        for (int r = 0; r < 8; ++r)
          if (ebase + r < E)
            __builtin_nontemporal_store(d[r] + b2v[t],
                                        outbase + (size_t)r * 128 + t * 16);
      }
    }
  }
}

extern "C" void kernel_launch(void* const* d_in, const int* in_sizes, int n_in,
                              void* d_out, int out_size, void* d_ws, size_t ws_size,
                              hipStream_t stream) {
  // setup_inputs order: src, dest, edge_attr, u, batch, W1, b1, W2, b2
  const float* src   = (const float*)d_in[0];
  const float* dst   = (const float*)d_in[1];
  const float* eattr = (const float*)d_in[2];
  // d_in[3] = u (unused), d_in[4] = batch (unused)
  const float* W1 = (const float*)d_in[5];
  const float* b1 = (const float*)d_in[6];
  const float* W2 = (const float*)d_in[7];
  const float* b2 = (const float*)d_in[8];
  float* out = (float*)d_out;

  const int E = in_sizes[0] / 512;
  const int ntiles = (E + 15) / 16;
  int blocks = (ntiles + WAVES_PER_BLOCK - 1) / WAVES_PER_BLOCK;
  if (blocks > 512) blocks = 512;
  if (blocks < 1) blocks = 1;

  edge_mlp_kernel<<<blocks, BLOCK_THREADS, 0, stream>>>(
      src, dst, eattr, W1, b1, W2, b2, out, E);
}
